// DKF_40905268527433
// MI455X (gfx1250) — compile-verified, matrix operand order checked by reading, not run on hardware
//
#include <hip/hip_runtime.h>
#include <hip/hip_bf16.h>

// ---------------------------------------------------------------------------
// Deep Kalman Filter fused implementation for gfx1250 (MI455X).
// Persistent-block design: each block owns 16 batch rows, keeps f16 WMMA
// fragment-permuted weights + recurrent state in LDS, loops over T=256 steps.
// All GEMMs via v_wmma_f32_16x16x32_f16; f16 sequence streams (h1/hidden_all)
// are prefetched one timestep ahead with the Tensor Data Mover
// (tensor_load_to_lds + s_wait_tensorcnt) into double-buffered LDS tiles.
// Transcendentals on the recurrent critical path use v_exp_f32/v_rcp_f32.
// ---------------------------------------------------------------------------

#define B_N 1024
#define T_N 256

typedef __attribute__((ext_vector_type(16))) _Float16 v16h;
typedef __attribute__((ext_vector_type(8)))  _Float16 v8h;
typedef __attribute__((ext_vector_type(8)))  float    v8f;
typedef __attribute__((ext_vector_type(4)))  unsigned int u32x4;
typedef __attribute__((ext_vector_type(8)))  int          i32x8;
typedef __attribute__((ext_vector_type(4)))  int          i32x4;

__device__ __forceinline__ float rcp_fast(float x)  { return __builtin_amdgcn_rcpf(x); }
__device__ __forceinline__ float sigf(float x)      { return rcp_fast(1.0f + __expf(-x)); }
__device__ __forceinline__ float tanh_fast(float x) { return 1.0f - 2.0f * rcp_fast(__expf(2.0f * x) + 1.0f); }
__device__ __forceinline__ float softplusf(float x) { return (x > 20.0f) ? x : log1pf(__expf(x)); }

__device__ __forceinline__ unsigned lds_addr_of(const void* p) {
  // generic shared-space address: low 32 bits are the LDS byte offset
  return (unsigned)(unsigned long long)p;
}

// ---------------------------------------------------------------------------
// Tensor Data Mover: async 2D f16 tile load (rows x rowlen) global -> LDS.
// LDS layout: each 128B row followed by 64B pad -> 96-half row stride.
// pad_interval=4 (16 qwords = 128B), pad_amount=15 (16 dwords = 64B).
// Descriptor per CDNA5 ISA ch.8 (D# group0 128b, group1 256b; groups 2/3 zero).
// ---------------------------------------------------------------------------
__device__ __forceinline__ void tdm_load_tile_f16(unsigned lds_byte_addr,
                                                  const _Float16* gptr,
                                                  int rows, int rowlen,
                                                  long row_stride_elems) {
  const unsigned long long ga = (unsigned long long)(size_t)gptr;
  u32x4 g0;
  g0[0] = 1u;                                               // count=1 (valid user D#)
  g0[1] = lds_byte_addr;                                    // lds_addr
  g0[2] = (unsigned)(ga & 0xFFFFFFFFu);                     // global_addr[31:0]
  g0[3] = (unsigned)((ga >> 32) & 0x01FFFFFFu) | (2u << 30);// global_addr[56:32], type=2
  i32x8 g1;
  g1[0] = (int)((1u << 16)            // data_size = 2 bytes
              | (1u << 20)            // pad_enable
              | (4u << 22)            // pad_interval: 2^4 qwords = 128B
              | (15u << 25));         // pad_amount: 16 dwords = 64B
  g1[1] = (int)(((unsigned)rowlen & 0xFFFFu) << 16);        // tensor_dim0[15:0]
  g1[2] = (int)((((unsigned)rowlen >> 16) & 0xFFFFu)        // tensor_dim0[31:16]
              | (((unsigned)rows & 0xFFFFu) << 16));        // tensor_dim1[15:0]
  g1[3] = (int)((((unsigned)rows >> 16) & 0xFFFFu)          // tensor_dim1[31:16]
              | (((unsigned)rowlen & 0xFFFFu) << 16));      // tile_dim0
  g1[4] = (int)((unsigned)rows & 0xFFFFu);                  // tile_dim1 (tile_dim2=0)
  const unsigned long long st = (unsigned long long)row_stride_elems;
  g1[5] = (int)(st & 0xFFFFFFFFu);                          // tensor_dim0_stride[31:0]
  g1[6] = (int)((st >> 32) & 0xFFFFu);                      // tensor_dim0_stride[47:32]
  g1[7] = 0;
  const i32x4 z4 = {0, 0, 0, 0};
  const i32x8 z8 = {0, 0, 0, 0, 0, 0, 0, 0};
  __builtin_amdgcn_tensor_load_to_lds(g0, g1, z4, z4, z8, 0);
}

// A-fragment (16x32 f16) from LDS row-major [16][stride]:
// lanes 0-15: M=lane, K={0..7,16..23}; lanes 16-31: M=lane-16, K={8..15,24..31}
__device__ __forceinline__ v16h load_a_frag(const _Float16* u, int stride, int kc) {
  const int lane = threadIdx.x & 31;
  const int m   = lane & 15;
  const int klo = (lane & 16) ? 8 : 0;
  const _Float16* p = u + m * stride + kc + klo;
  union { v16h v; v8h h[2]; } r;
  r.h[0] = *(const v8h*)(p);
  r.h[1] = *(const v8h*)(p + 16);
  return r.v;
}

// B-fragment from a pre-permuted LDS buffer: [chunk][tile][lane][16] contiguous.
__device__ __forceinline__ v16h load_b_frag(const _Float16* fragbuf, int chunk, int tile, int ntiles) {
  const int lane = threadIdx.x & 31;
  const _Float16* p = fragbuf + (((chunk * ntiles + tile) * 32 + lane) << 4);
  union { v16h v; v8h h[2]; } r;
  r.h[0] = *(const v8h*)(p);
  r.h[1] = *(const v8h*)(p + 8);
  return r.v;
}

// C/D layout: VGPR r holds M=r (lanes 0-15), M=r+8 (lanes 16-31), N=lane&15.
__device__ __forceinline__ void store_c_frag(float* z, int stride, int ncol0, v8f c) {
  const int lane = threadIdx.x & 31;
  const int n  = ncol0 + (lane & 15);
  const int mb = (lane & 16) ? 8 : 0;
#pragma unroll
  for (int r = 0; r < 8; ++r) z[(mb + r) * stride + n] = c[r];
}

template <int KCH>
__device__ __forceinline__ void wmma_gemm(const _Float16* A, int strideA,
                                          const _Float16* bfrag, int ntiles,
                                          float* z, int strideZ) {
  const int wave = threadIdx.x >> 5;
  if (wave < ntiles) {  // wave-uniform branch: EXEC stays all-ones for WMMA
    v8f acc = {};
#pragma unroll
    for (int c = 0; c < KCH; ++c) {
      v16h a = load_a_frag(A, strideA, c * 32);
      v16h b = load_b_frag(bfrag, c, wave, ntiles);
      acc = __builtin_amdgcn_wmma_f32_16x16x32_f16(false, a, false, b, (short)0, acc, false, false);
    }
    store_c_frag(z, strideZ, wave * 16, acc);
  }
}

// Permute a row-major [K][N] f32 weight into WMMA B fragments (f16) in LDS.
__device__ __forceinline__ void stage_knw(_Float16* dst, const float* src,
                                          int K, int N, int kch, int ntiles) {
  const int total = kch * ntiles * 512;
  for (int e = threadIdx.x; e < total; e += blockDim.x) {
    int j = e & 15, lane = (e >> 4) & 31, tw = e >> 9;
    int tile = tw % ntiles, chunk = tw / ntiles;
    int n   = tile * 16 + (lane & 15);
    int klo = (lane & 16) ? 8 : 0;
    int k   = chunk * 32 + klo + (j < 8 ? j : j + 8);
    dst[e] = (_Float16)((k < K) ? src[(size_t)k * N + n] : 0.0f);
  }
}

// Two-barrier block reduction: wave32 shuffle butterflies + cross-wave LDS.
// Result valid in thread 0.
__device__ __forceinline__ float block_reduce(float v, float* red8) {
  const int lane = threadIdx.x & 31, wave = threadIdx.x >> 5;
#pragma unroll
  for (int o = 16; o > 0; o >>= 1) v += __shfl_down(v, o, 32);
  if (lane == 0) red8[wave] = v;
  __syncthreads();
  float s = 0.0f;
  if (threadIdx.x == 0) {
#pragma unroll
    for (int i = 0; i < 8; ++i) s += red8[i];
  }
  __syncthreads();   // protect red8 for reuse
  return s;
}

// ---------------------------------------------------------------------------
// LSTM, DIN=16 (rec LSTM + bilstm layer0 fwd/bwd).  Kpad=64, KCH=2, NT=8.
// blockIdx.y: 0 = rec (out: hT fp32 at t=T-1), 1 = l0 fwd (h1 cols 0..31),
//             2 = l0 bwd (h1 cols 32..63, reversed time).
// f32 observations converted to f16 on load; next step prefetched.
// ---------------------------------------------------------------------------
__global__ __launch_bounds__(256)
void lstm16_kernel(const float* __restrict__ obs,
                   const float* WihA, const float* WhhA, const float* bA,
                   const float* WihB, const float* WhhB, const float* bB,
                   const float* WihC, const float* WhhC, const float* bC,
                   float* __restrict__ hT, _Float16* __restrict__ h1) {
  __shared__ __align__(16) _Float16 u[16 * 64];          // [x(16) | h(32) | pad(16)]
  __shared__ __align__(16) _Float16 bfrag[2 * 8 * 512];  // 16 KB
  __shared__ float zb[16 * 128];
  __shared__ float cbuf[512];
  __shared__ float bb[128];

  const int tid     = threadIdx.x;
  const int variant = blockIdx.y;
  const int b0      = blockIdx.x * 16;
  const float* Wih  = (variant == 0) ? WihA : (variant == 1) ? WihB : WihC;
  const float* Whh  = (variant == 0) ? WhhA : (variant == 1) ? WhhB : WhhC;
  const float* bias = (variant == 0) ? bA   : (variant == 1) ? bB   : bC;
  const bool rev = (variant == 2);

  for (int e = tid; e < 2 * 8 * 512; e += 256) {
    int j = e & 15, lane = (e >> 4) & 31, tw = e >> 9;
    int tile = tw & 7, chunk = tw >> 3;
    int n = tile * 16 + (lane & 15);
    int klo = (lane & 16) ? 8 : 0;
    int k = chunk * 32 + klo + (j < 8 ? j : j + 8);
    float w = 0.0f;
    if (k < 16)      w = Wih[n * 16 + k];
    else if (k < 48) w = Whh[n * 32 + (k - 16)];
    bfrag[e] = (_Float16)w;
  }
  for (int i = tid; i < 128; i += 256) bb[i] = bias[i];
  for (int e = tid; e < 1024; e += 256) u[e] = (_Float16)0.0f;
  for (int e = tid; e < 512;  e += 256) cbuf[e] = 0.0f;
  __syncthreads();

  for (int s = 0; s < T_N; ++s) {
    const int t = rev ? (T_N - 1 - s) : s;
    {
      const int r = tid >> 4, d = tid & 15;
      const float* src = obs + ((size_t)(b0 + r) * T_N + t) * 16 + d;
      u[r * 64 + d] = (_Float16)(*src);
      if (s + 1 < T_N) __builtin_prefetch(src + (rev ? -16 : 16), 0, 1);
    }
    __syncthreads();
    wmma_gemm<2>(u, 64, bfrag, 8, zb, 128);
    __syncthreads();
    for (int e = tid; e < 512; e += 256) {
      const int r = e >> 5, he = e & 31;
      const float zi = zb[r * 128 + he]       + bb[he];
      const float zf = zb[r * 128 + 32 + he]  + bb[32 + he];
      const float zg = zb[r * 128 + 64 + he]  + bb[64 + he];
      const float zo = zb[r * 128 + 96 + he]  + bb[96 + he];
      const float c  = sigf(zf) * cbuf[e] + sigf(zi) * tanh_fast(zg);
      cbuf[e] = c;
      const float h = sigf(zo) * tanh_fast(c);
      u[r * 64 + 16 + he] = (_Float16)h;
      if (variant == 0) {
        if (t == T_N - 1) hT[(size_t)(b0 + r) * 32 + he] = h;
      } else {
        h1[((size_t)(b0 + r) * T_N + t) * 64 + ((variant == 1) ? 0 : 32) + he] = (_Float16)h;
      }
    }
    __syncthreads();
  }
}

// ---------------------------------------------------------------------------
// LSTM, DIN=64 (bilstm layer1 fwd/bwd over h1 f16).  Kpad=96, KCH=3, NT=8.
// Double-buffered input tiles streamed one step ahead via TDM.
// blockIdx.y: 0 = fwd (hidden_all cols 0..31), 1 = bwd (cols 32..63).
// ---------------------------------------------------------------------------
__global__ __launch_bounds__(256)
void lstm64_kernel(const _Float16* __restrict__ xin,
                   const float* WihA, const float* WhhA, const float* bA,
                   const float* WihB, const float* WhhB, const float* bB,
                   _Float16* __restrict__ hout) {
  __shared__ __align__(16) _Float16 u2[2 * 16 * 96];     // ping-pong [x(64)|h(32)]
  __shared__ __align__(16) _Float16 bfrag[3 * 8 * 512];  // 24 KB
  __shared__ float zb[16 * 128];
  __shared__ float cbuf[512];
  __shared__ float bb[128];

  const int tid     = threadIdx.x;
  const int wave    = tid >> 5;
  const int variant = blockIdx.y;
  const int b0      = blockIdx.x * 16;
  const float* Wih  = (variant == 0) ? WihA : WihB;
  const float* Whh  = (variant == 0) ? WhhA : WhhB;
  const float* bias = (variant == 0) ? bA   : bB;
  const bool rev = (variant == 1);

  for (int e = tid; e < 3 * 8 * 512; e += 256) {
    int j = e & 15, lane = (e >> 4) & 31, tw = e >> 9;
    int tile = tw & 7, chunk = tw >> 3;
    int n = tile * 16 + (lane & 15);
    int klo = (lane & 16) ? 8 : 0;
    int k = chunk * 32 + klo + (j < 8 ? j : j + 8);
    float w = 0.0f;
    if (k < 64)      w = Wih[n * 64 + k];
    else if (k < 96) w = Whh[n * 32 + (k - 64)];
    bfrag[e] = (_Float16)w;
  }
  for (int i = tid; i < 128;  i += 256) bb[i] = bias[i];
  for (int e = tid; e < 3072; e += 256) u2[e] = (_Float16)0.0f;
  for (int e = tid; e < 512;  e += 256) cbuf[e] = 0.0f;
  __syncthreads();

  // kick off TDM load of the first input tile into buffer 0
  if (wave == 0) {
    const int t0 = rev ? (T_N - 1) : 0;
    tdm_load_tile_f16(lds_addr_of(u2),
                      xin + ((size_t)b0 * T_N + t0) * 64,
                      16, 64, (long)T_N * 64);
  }

  for (int s = 0; s < T_N; ++s) {
    const int t   = rev ? (T_N - 1 - s) : s;
    const int act = s & 1, nxt = act ^ 1;
    if (wave == 0) {
      // prefetch next step's tile (wraps at the end: harmless reload of step 0)
      const int sn = (s + 1) & (T_N - 1);
      const int tn = rev ? (T_N - 1 - sn) : sn;
      tdm_load_tile_f16(lds_addr_of(u2 + nxt * 1536),
                        xin + ((size_t)b0 * T_N + tn) * 64,
                        16, 64, (long)T_N * 64);
      __builtin_amdgcn_s_wait_tensorcnt((short)1);  // current tile resident
    }
    __syncthreads();
    wmma_gemm<3>(u2 + act * 1536, 96, bfrag, 8, zb, 128);
    __syncthreads();
    for (int e = tid; e < 512; e += 256) {
      const int r = e >> 5, he = e & 31;
      const float zi = zb[r * 128 + he]       + bb[he];
      const float zf = zb[r * 128 + 32 + he]  + bb[32 + he];
      const float zg = zb[r * 128 + 64 + he]  + bb[64 + he];
      const float zo = zb[r * 128 + 96 + he]  + bb[96 + he];
      const float c  = sigf(zf) * cbuf[e] + sigf(zi) * tanh_fast(zg);
      cbuf[e] = c;
      const float h = sigf(zo) * tanh_fast(c);
      u2[nxt * 1536 + r * 96 + 64 + he] = (_Float16)h;   // recurrent input for s+1
      hout[((size_t)(b0 + r) * T_N + t) * 64 + ((variant == 0) ? 0 : 32) + he] = (_Float16)h;
    }
    __syncthreads();
  }
  if (wave == 0) __builtin_amdgcn_s_wait_tensorcnt((short)0);
}

// ---------------------------------------------------------------------------
// Fused scan: x0 init (+qm0_KL partial), then T steps of trans-MLP + posterior
// net, ELBO partials, mean/var outputs.  Each block: 16 batch rows.
// hidden_all tiles streamed one step ahead via TDM into ping-pong buffers.
// ---------------------------------------------------------------------------
__global__ __launch_bounds__(256)
void scan_kernel(const float* __restrict__ obs, const float* __restrict__ eps0,
                 const float* __restrict__ eps_seq,
                 const float* Hm, const float* qvar, const float* rvar,
                 const float* tW1, const float* tb1, const float* tW2, const float* tb2,
                 const float* tW3, const float* tb3, const float* tW4, const float* tb4,
                 const float* rWm, const float* rbm, const float* rWv, const float* rbv,
                 const float* pW1, const float* pb1, const float* pWm, const float* pbm,
                 const float* pWv, const float* pbv,
                 const float* __restrict__ hT, const _Float16* __restrict__ hidden_all,
                 float* __restrict__ out, float* __restrict__ acc) {
  __shared__ __align__(16) _Float16 wf1[1 * 8 * 512];   // trans W1  32x128
  __shared__ __align__(16) _Float16 wf2[4 * 4 * 512];   // trans W2 128x64
  __shared__ __align__(16) _Float16 wf3[2 * 2 * 512];   // trans W3  64x32
  __shared__ __align__(16) _Float16 wf4[1 * 2 * 512];   // trans W4  32x32
  __shared__ __align__(16) _Float16 wp1[3 * 4 * 512];   // post  W1  96x64
  __shared__ __align__(16) _Float16 wpm[2 * 2 * 512];   // post  Wm  64x32
  __shared__ __align__(16) _Float16 wpv[2 * 2 * 512];   // post  Wv  64x32
  __shared__ float bL[384];   // b1@0 b2@128 b3@192 b4@224 pb1@256 pbm@320 pbv@352
  __shared__ float HL[512], H2L[512];
  __shared__ float rvinvL[16], l2prL[16], qvinvL[32], lqvL[32];
  __shared__ __align__(16) _Float16 xbuf[512];          // x_prev / x_post, 16x32
  __shared__ __align__(16) _Float16 a1[2048];           // 16x128
  __shared__ __align__(16) _Float16 a2[1024];           // 16x64
  __shared__ __align__(16) _Float16 a3[512];            // 16x32
  __shared__ __align__(16) _Float16 pc2[2 * 1536];      // ping-pong [x_t | h_t], 16x96
  __shared__ __align__(16) _Float16 ph[1024];           // 16x64
  __shared__ float xt32[512], mbuf[512], vbuf[512];
  __shared__ float zb[2048];                            // GEMM f32 output scratch
  __shared__ float red8[8];

  const int tid  = threadIdx.x;
  const int wave = tid >> 5;
  const int b0   = blockIdx.x * 16;

  stage_knw(wf1, tW1,  32, 128, 1, 8);
  stage_knw(wf2, tW2, 128,  64, 4, 4);
  stage_knw(wf3, tW3,  64,  32, 2, 2);
  stage_knw(wf4, tW4,  32,  32, 1, 2);
  stage_knw(wp1, pW1,  96,  64, 3, 4);
  stage_knw(wpm, pWm,  64,  32, 2, 2);
  stage_knw(wpv, pWv,  64,  32, 2, 2);
  for (int i = tid; i < 128; i += 256) bL[i] = tb1[i];
  for (int i = tid; i < 64;  i += 256) { bL[128 + i] = tb2[i]; bL[256 + i] = pb1[i]; }
  for (int i = tid; i < 32;  i += 256) {
    bL[192 + i] = tb3[i]; bL[224 + i] = tb4[i]; bL[320 + i] = pbm[i]; bL[352 + i] = pbv[i];
    qvinvL[i] = 1.0f / qvar[i]; lqvL[i] = logf(qvar[i]);
  }
  for (int i = tid; i < 16;  i += 256) {
    rvinvL[i] = 1.0f / rvar[i]; l2prL[i] = logf(6.2831853071795864f * rvar[i]);
  }
  for (int i = tid; i < 512; i += 256) { const float h = Hm[i]; HL[i] = h; H2L[i] = h * h; }
  for (int e = tid; e < 512; e += 256) zb[e] = hT[(size_t)b0 * 32 + e];

  // kick off TDM load of h_0 into buffer 0 (cols 32..95 of pc2[0])
  if (wave == 0) {
    tdm_load_tile_f16(lds_addr_of(pc2 + 32),
                      hidden_all + (size_t)b0 * T_N * 64,
                      16, 64, (long)T_N * 64);
  }
  __syncthreads();

  // ---- init: m0/v0/x0, qm0_KL partial ----
  float part = 0.0f;
  for (int e = tid; e < 512; e += 256) {
    const int r = e >> 5, d = e & 31;
    float m0 = rbm[d], zv = rbv[d];
    for (int k = 0; k < 32; ++k) {
      const float hk = zb[r * 32 + k];
      m0 += hk * rWm[k * 32 + d];
      zv += hk * rWv[k * 32 + d];
    }
    const float v0 = softplusf(zv) + 1e-6f;
    part += v0 + m0 * m0 - 1.0f - logf(v0);
    xbuf[r * 32 + d] = (_Float16)(m0 + sqrtf(v0) * eps0[(size_t)b0 * 32 + e]);
  }
  {
    const float s = block_reduce(part, red8);
    if (tid == 0) atomicAdd(acc + 0, s);
  }
  __syncthreads();

  float* const out_m = out + 1;
  float* const out_v = out + 1 + (size_t)B_N * T_N * 32;

  float df_acc = 0.0f;
  for (int t = 0; t < T_N; ++t) {
    const int act = t & 1, nxt = act ^ 1;
    if (wave == 0) {
      const int tn = (t + 1) & (T_N - 1);  // wrap: harmless reload at the end
      tdm_load_tile_f16(lds_addr_of(pc2 + nxt * 1536 + 32),
                        hidden_all + ((size_t)b0 * T_N + tn) * 64,
                        16, 64, (long)T_N * 64);
    }

    // trans MLP: 32 -> 128 -> 64 -> 32 -> 32
    wmma_gemm<1>(xbuf, 32, wf1, 8, zb, 128);
    __syncthreads();
    for (int e = tid; e < 2048; e += 256) a1[e] = (_Float16)fmaxf(zb[e] + bL[e & 127], 0.0f);
    __syncthreads();
    wmma_gemm<4>(a1, 128, wf2, 4, zb, 64);
    __syncthreads();
    for (int e = tid; e < 1024; e += 256) a2[e] = (_Float16)fmaxf(zb[e] + bL[128 + (e & 63)], 0.0f);
    __syncthreads();
    wmma_gemm<2>(a2, 64, wf3, 2, zb, 32);
    __syncthreads();
    for (int e = tid; e < 512; e += 256) a3[e] = (_Float16)fmaxf(zb[e] + bL[192 + (e & 31)], 0.0f);
    __syncthreads();
    wmma_gemm<1>(a3, 32, wf4, 2, zb, 32);
    __syncthreads();
    for (int e = tid; e < 512; e += 256) {
      const int r = e >> 5, d = e & 31;
      const float xt = zb[e] + bL[224 + d];
      xt32[e] = xt;
      pc2[act * 1536 + r * 96 + d] = (_Float16)xt;
    }
    if (wave == 0) __builtin_amdgcn_s_wait_tensorcnt((short)1);  // h_t resident
    __syncthreads();

    // posterior: relu(concat @ W1 + b1), then m / v in parallel across waves
    wmma_gemm<3>(pc2 + act * 1536, 96, wp1, 4, zb, 64);
    __syncthreads();
    for (int e = tid; e < 1024; e += 256) ph[e] = (_Float16)fmaxf(zb[e] + bL[256 + (e & 63)], 0.0f);
    __syncthreads();
    if (wave < 4) {
      const _Float16* fb = (wave < 2) ? wpm : wpv;
      const int tile = wave & 1;
      v8f accv = {};
#pragma unroll
      for (int c = 0; c < 2; ++c) {
        v16h a = load_a_frag(ph, 64, c * 32);
        v16h b = load_b_frag(fb, c, tile, 2);
        accv = __builtin_amdgcn_wmma_f32_16x16x32_f16(false, a, false, b, (short)0, accv, false, false);
      }
      store_c_frag(zb, 64, ((wave < 2) ? 0 : 32) + tile * 16, accv);
    }
    __syncthreads();
    for (int e = tid; e < 512; e += 256) {
      const int r = e >> 5, d = e & 31;
      const float m = zb[r * 64 + d] + bL[320 + d];
      const float v = softplusf(zb[r * 64 + 32 + d] + bL[352 + d]) + 1e-6f;
      mbuf[e] = m; vbuf[e] = v;
      const size_t oidx = ((size_t)(b0 + r) * T_N + t) * 32 + d;
      out_m[oidx] = m;
      out_v[oidx] = v;
      xbuf[r * 32 + d] = (_Float16)(m + sqrtf(v) * eps_seq[oidx]);
    }
    __syncthreads();

    // data-fit partial: -0.5 * sum_i ((y-cm)^2/r + log(2*pi*r) + cvc/r)
    float part2;
    {
      const int r = tid >> 4, i = tid & 15;
      float cm = 0.0f, cvc = 0.0f;
      for (int d = 0; d < 32; ++d) {
        cm  += mbuf[r * 32 + d] * HL[i * 32 + d];
        cvc += vbuf[r * 32 + d] * H2L[i * 32 + d];
      }
      const float y  = obs[((size_t)(b0 + r) * T_N + t) * 16 + i];
      const float dy = y - cm;
      part2 = -0.5f * ((dy * dy + cvc) * rvinvL[i] + l2prL[i]);
    }
    {
      const float s = block_reduce(part2, red8);
      if (tid == 0) df_acc += s;
    }

    if (t == T_N - 1) {  // only kls[-1] contributes to the ELBO
      float klp = 0.0f;
      for (int e = tid; e < 512; e += 256) {
        const int d = e & 31;
        const float v = vbuf[e], dm = mbuf[e] - xt32[e];
        klp += lqvL[d] - logf(v) + (v + dm * dm) * qvinvL[d] - 1.0f;
      }
      const float s = block_reduce(klp, red8);
      if (tid == 0) atomicAdd(acc + 2, 0.5f * s);
    }
    __syncthreads();
  }
  if (tid == 0) atomicAdd(acc + 1, df_acc);
  if (wave == 0) __builtin_amdgcn_s_wait_tensorcnt((short)0);
}

__global__ void init_acc_kernel(float* acc) {
  if (threadIdx.x < 4) acc[threadIdx.x] = 0.0f;
}

__global__ void finalize_kernel(const float* __restrict__ acc, float* __restrict__ out) {
  if (threadIdx.x == 0) {
    const float qm0 = 0.5f * acc[0] / (float)B_N;
    const float lik = acc[1] / ((float)B_N * (float)T_N);
    const float kld = (acc[2] / (float)B_N) * (1.0f + 1.0f / (float)T_N);
    out[0] = -qm0 + lik - kld;
  }
}

// ---------------------------------------------------------------------------
// Input index map (setup_inputs dict order, nested dicts flattened in order)
// ---------------------------------------------------------------------------
enum {
  I_OBS = 0, I_EPS0, I_EPSSEQ, I_H, I_QVAR, I_RVAR,
  T_W1, T_B1, T_W2, T_B2, T_W3, T_B3, T_W4, T_B4,
  R_WIH, R_WHH, R_B, R_WM, R_BM, R_WV, R_BV,
  E_WIH0F, E_WHH0F, E_B0F, E_WIH0B, E_WHH0B, E_B0B,
  E_WIH1F, E_WHH1F, E_B1F, E_WIH1B, E_WHH1B, E_B1B,
  P_W1, P_B1, P_WM, P_BM, P_WV, P_BV
};

extern "C" void kernel_launch(void* const* d_in, const int* in_sizes, int n_in,
                              void* d_out, int out_size, void* d_ws, size_t ws_size,
                              hipStream_t stream) {
  (void)in_sizes; (void)n_in; (void)out_size; (void)ws_size;
#define FIN(i) ((const float*)d_in[(i)])

  char* ws = (char*)d_ws;
  float*     acc  = (float*)ws;                                   // 4 floats (+pad)
  float*     hT   = (float*)(ws + 256);                           // B*32 f32   = 128 KB
  _Float16*  h1   = (_Float16*)(ws + 256 + (size_t)B_N * 32 * 4); // B*T*64 f16 = 32 MB
  _Float16*  hall = (_Float16*)((char*)h1 + (size_t)B_N * T_N * 64 * 2);
  float*     outp = (float*)d_out;

  init_acc_kernel<<<1, 32, 0, stream>>>(acc);

  lstm16_kernel<<<dim3(B_N / 16, 3), 256, 0, stream>>>(
      FIN(I_OBS),
      FIN(R_WIH),   FIN(R_WHH),   FIN(R_B),
      FIN(E_WIH0F), FIN(E_WHH0F), FIN(E_B0F),
      FIN(E_WIH0B), FIN(E_WHH0B), FIN(E_B0B),
      hT, h1);

  lstm64_kernel<<<dim3(B_N / 16, 2), 256, 0, stream>>>(
      h1,
      FIN(E_WIH1F), FIN(E_WHH1F), FIN(E_B1F),
      FIN(E_WIH1B), FIN(E_WHH1B), FIN(E_B1B),
      hall);

  scan_kernel<<<B_N / 16, 256, 0, stream>>>(
      FIN(I_OBS), FIN(I_EPS0), FIN(I_EPSSEQ),
      FIN(I_H), FIN(I_QVAR), FIN(I_RVAR),
      FIN(T_W1), FIN(T_B1), FIN(T_W2), FIN(T_B2),
      FIN(T_W3), FIN(T_B3), FIN(T_W4), FIN(T_B4),
      FIN(R_WM), FIN(R_BM), FIN(R_WV), FIN(R_BV),
      FIN(P_W1), FIN(P_B1), FIN(P_WM), FIN(P_BM),
      FIN(P_WV), FIN(P_BV),
      hT, hall, outp, acc);

  finalize_kernel<<<1, 32, 0, stream>>>(acc, outp);
#undef FIN
}
